// MultiHeadSelfAttention_45492293599620
// MI455X (gfx1250) — matrix operand order repaired and checked by reading, not compiled, not run on hardware
//
#include <hip/hip_runtime.h>
#include <cstdint>
#include <cstddef>

// ---------------------------------------------------------------- types
typedef _Float16 f16;
typedef __attribute__((ext_vector_type(16))) _Float16 v16h;
typedef __attribute__((ext_vector_type(8)))  _Float16 v8h;
typedef __attribute__((ext_vector_type(8)))  float    v8f;

#define Bc 2
#define Sc 2048
#define Dc 1024
#define Hc 16
#define HDc 64
#define BSc (Bc*Sc)
#define SCALEc 0.125f

__device__ __forceinline__ v16h cat8(v8h lo, v8h hi) {
  return __builtin_shufflevector(lo, hi, 0, 1, 2, 3, 4, 5, 6, 7, 8, 9, 10, 11, 12, 13, 14, 15);
}

// ------------------------------------------------ WMMA fragment loaders
// A fragment 16x32 (MxK), memory row-major [m][k], p[m*ld + k].
// Per lane: halves  k = 8g..8g+7  and  k = 16+8g..16+8g+7  (two b128 loads).
__device__ __forceinline__ v16h frag_a16(const f16* p, int ld) {
  const int lane = threadIdx.x & 31;
  const int m = lane & 15, g = lane >> 4;
  const v8h lo = *(const v8h*)(p + m * ld + 8 * g);
  const v8h hi = *(const v8h*)(p + m * ld + 8 * g + 16);
  return cat8(lo, hi);
}

// B fragment 32x16 (KxN) where memory holds the TRANSPOSE [n][k]: element (k,n) at p[n*ld + k].
// Per lane: halves  k = 8g..8g+7  and  k = 16+8g..16+8g+7  (two b128 loads).
__device__ __forceinline__ v16h frag_bt16(const f16* p, int ld) {
  const int lane = threadIdx.x & 31;
  const int n = lane & 15, g = lane >> 4;
  const v8h lo = *(const v8h*)(p + n * ld + 8 * g);
  const v8h hi = *(const v8h*)(p + n * ld + 8 * g + 16);
  return cat8(lo, hi);
}

#define WMMA(a, b, c) \
  __builtin_amdgcn_wmma_f32_16x16x32_f16(false, (a), false, (b), (short)0, (c), false, false)

// ---------------------------------------------------------------- utils
// Transposing convert: dst[n*D + k] = (f16) src[k*D + n]
__global__ __launch_bounds__(256) void cvt_t_f32_f16(const float* __restrict__ src,
                                                     f16* __restrict__ dst, int n) {
  int i = blockIdx.x * 256 + threadIdx.x;
  if (i < n) {
    const int nn = i >> 10, kk = i & (Dc - 1);
    dst[i] = (f16)src[kk * Dc + nn];
  }
}

// One block per row (1024 elems). mode 0: out16 = f16(LN(x)); mode 1: out32 = LN(x+add)
__global__ __launch_bounds__(256) void layernorm_k(const float* __restrict__ x,
                                                   const float* __restrict__ add,
                                                   const float* __restrict__ gam,
                                                   const float* __restrict__ bet,
                                                   f16* __restrict__ out16,
                                                   float* __restrict__ out32, int mode) {
  __shared__ float red[256];
  const int tid = threadIdx.x;
  const size_t base = (size_t)blockIdx.x * Dc;
  float v[4];
  float s = 0.f;
#pragma unroll
  for (int j = 0; j < 4; ++j) {
    const int d = tid + 256 * j;
    float t = x[base + d];
    if (mode) t += add[base + d];
    v[j] = t;
    s += t;
  }
  red[tid] = s; __syncthreads();
  for (int st = 128; st > 0; st >>= 1) { if (tid < st) red[tid] += red[tid + st]; __syncthreads(); }
  const float mu = red[0] * (1.f / Dc);
  __syncthreads();
  float q = 0.f;
#pragma unroll
  for (int j = 0; j < 4; ++j) { const float d = v[j] - mu; q += d * d; }
  red[tid] = q; __syncthreads();
  for (int st = 128; st > 0; st >>= 1) { if (tid < st) red[tid] += red[tid + st]; __syncthreads(); }
  const float inv = rsqrtf(red[0] * (1.f / Dc) + 1e-5f);
#pragma unroll
  for (int j = 0; j < 4; ++j) {
    const int d = tid + 256 * j;
    const float y = (v[j] - mu) * inv * gam[d] + bet[d];
    if (mode) out32[base + d] = y; else out16[base + d] = (f16)y;
  }
}

// --------------------------------------------- QKV projection GEMM (WMMA)
// grid (D/64, BS/256, 3); 256 thr = 8 waves; wave tile = 32 rows x 64 cols (8 WMMAs/K-step).
// Weights pre-transposed WT[n][k]. q/k written [B,H,S,HD]; v written [B,H,HD,S].
__global__ __launch_bounds__(256) void gemm_qkv(const f16* __restrict__ h16,
    const f16* __restrict__ WqT, const f16* __restrict__ WkT, const f16* __restrict__ WvT,
    const float* __restrict__ bq, const float* __restrict__ bk, const float* __restrict__ bv,
    f16* __restrict__ qo, f16* __restrict__ ko, f16* __restrict__ vo) {
  const int z = blockIdx.z;
  const f16*   WT   = (z == 0) ? WqT : (z == 1) ? WkT : WvT;
  const float* bias = (z == 0) ? bq : (z == 1) ? bk : bv;
  f16*         out  = (z == 0) ? qo : (z == 1) ? ko : vo;
  const int head = blockIdx.x;
  const int n0 = head * 64;
  const int r0 = blockIdx.y * 256;
  __shared__ __attribute__((aligned(16))) f16 As[256 * 40];   // [row][k], 80B pitch
  __shared__ __attribute__((aligned(16))) f16 BTs[64 * 40];   // [n][k]
  const int tid = threadIdx.x, w = tid >> 5, lane = tid & 31;
  const int ln = lane & 15, g = lane >> 4;
  const int ar = tid >> 2, ac = (tid & 3) * 8;   // fill coords (exact fit, no predication)
  v8f acc[8] = {};
  for (int k0 = 0; k0 < Dc; k0 += 32) {
    __syncthreads();
#pragma unroll
    for (int u = 0; u < 4; ++u)                   // A: 256 rows x 4 chunks
      *(v8h*)&As[(ar + 64 * u) * 40 + ac] =
          *(const v8h*)&h16[(size_t)(r0 + ar + 64 * u) * Dc + k0 + ac];
    *(v8h*)&BTs[(ar & 63) * 40 + ac + ((ar >> 6) ? 0 : 0)] =
        *(const v8h*)&WT[(size_t)(n0 + (ar & 63)) * Dc + k0 + ac];
    if (k0 + 32 < Dc) {
      __builtin_prefetch(&h16[(size_t)(r0 + ar) * Dc + k0 + 32 + ac], 0, 1);
      __builtin_prefetch(&WT[(size_t)(n0 + (ar & 63)) * Dc + k0 + 32 + ac], 0, 1);
    }
    __syncthreads();
    v16h bfr[4];
#pragma unroll
    for (int t = 0; t < 4; ++t) bfr[t] = frag_bt16(&BTs[(16 * t) * 40], 40);
    const v16h a0 = frag_a16(&As[(32 * w) * 40], 40);
    const v16h a1 = frag_a16(&As[(32 * w + 16) * 40], 40);
#pragma unroll
    for (int t = 0; t < 4; ++t) acc[t] = WMMA(a0, bfr[t], acc[t]);
#pragma unroll
    for (int t = 0; t < 4; ++t) acc[4 + t] = WMMA(a1, bfr[t], acc[4 + t]);
  }
#pragma unroll
  for (int u = 0; u < 2; ++u)
#pragma unroll
    for (int t = 0; t < 4; ++t) {
      const int hd = 16 * t + ln;
      const float bb = bias[n0 + hd];
#pragma unroll
      for (int i = 0; i < 8; ++i) {
        const int r = r0 + 32 * w + 16 * u + 8 * g + i;
        const int b_ = r >> 11, s_ = r & (Sc - 1);
        const f16 val = (f16)(acc[4 * u + t][i] + bb);
        if (z == 2)
          out[(((size_t)b_ * Hc + head) * HDc + hd) * Sc + s_] = val;   // V: [B,H,HD,S]
        else
          out[(((size_t)b_ * Hc + head) * Sc + s_) * HDc + hd] = val;   // Q,K: [B,H,S,HD]
      }
    }
}

// --------------------------------------------- output projection GEMM (WMMA)
// ctx16 [BS,D] @ Wo (WoT pre-transposed) + bo -> proj f32 [BS,D]
__global__ __launch_bounds__(256) void gemm_o(const f16* __restrict__ ctx,
    const f16* __restrict__ WT, const float* __restrict__ bias, float* __restrict__ out) {
  const int n0 = blockIdx.x * 64;
  const int r0 = blockIdx.y * 256;
  __shared__ __attribute__((aligned(16))) f16 As[256 * 40];
  __shared__ __attribute__((aligned(16))) f16 BTs[64 * 40];
  const int tid = threadIdx.x, w = tid >> 5, lane = tid & 31;
  const int ln = lane & 15, g = lane >> 4;
  const int ar = tid >> 2, ac = (tid & 3) * 8;
  v8f acc[8] = {};
  for (int k0 = 0; k0 < Dc; k0 += 32) {
    __syncthreads();
#pragma unroll
    for (int u = 0; u < 4; ++u)
      *(v8h*)&As[(ar + 64 * u) * 40 + ac] =
          *(const v8h*)&ctx[(size_t)(r0 + ar + 64 * u) * Dc + k0 + ac];
    *(v8h*)&BTs[(ar & 63) * 40 + ac] =
        *(const v8h*)&WT[(size_t)(n0 + (ar & 63)) * Dc + k0 + ac];
    if (k0 + 32 < Dc) {
      __builtin_prefetch(&ctx[(size_t)(r0 + ar) * Dc + k0 + 32 + ac], 0, 1);
      __builtin_prefetch(&WT[(size_t)(n0 + (ar & 63)) * Dc + k0 + 32 + ac], 0, 1);
    }
    __syncthreads();
    v16h bfr[4];
#pragma unroll
    for (int t = 0; t < 4; ++t) bfr[t] = frag_bt16(&BTs[(16 * t) * 40], 40);
    const v16h a0 = frag_a16(&As[(32 * w) * 40], 40);
    const v16h a1 = frag_a16(&As[(32 * w + 16) * 40], 40);
#pragma unroll
    for (int t = 0; t < 4; ++t) acc[t] = WMMA(a0, bfr[t], acc[t]);
#pragma unroll
    for (int t = 0; t < 4; ++t) acc[4 + t] = WMMA(a1, bfr[t], acc[4 + t]);
  }
#pragma unroll
  for (int u = 0; u < 2; ++u)
#pragma unroll
    for (int t = 0; t < 4; ++t) {
      const int n = n0 + 16 * t + ln;
      const float bb = bias[n];
#pragma unroll
      for (int i = 0; i < 8; ++i) {
        const int r = r0 + 32 * w + 16 * u + 8 * g + i;
        out[(size_t)r * Dc + n] = acc[4 * u + t][i] + bb;
      }
    }
}

// --------------------------------------------- flash attention (WMMA, online softmax)
// grid (S/64, B*H); 128 thr = 4 waves; wave = 16 query rows.
// q,k: [B,H,S,HD]; v: [B,H,HD,S] (transposed so V-tiles load contiguously).
__global__ __launch_bounds__(128) void attention_k(const f16* __restrict__ q,
    const f16* __restrict__ k, const f16* __restrict__ v, f16* __restrict__ ctx) {
  const int bh = blockIdx.y;
  const int b = bh >> 4, h = bh & 15;
  const int q0 = blockIdx.x * 64;
  const int tid = threadIdx.x, w = tid >> 5, lane = tid & 31;
  const int ln = lane & 15, g = lane >> 4;
  __shared__ __attribute__((aligned(16))) f16 Ks[64 * 72];   // [key][hd], 144B pitch
  __shared__ __attribute__((aligned(16))) f16 Vt[64 * 72];   // [hd][key], 144B pitch
  __shared__ __attribute__((aligned(16))) f16 Ps[4][16 * 64];
  const size_t base  = (size_t)bh * Sc * HDc;   // q,k
  const size_t baseV = (size_t)bh * HDc * Sc;   // v (transposed)

  // Q fragments (global, contiguous b128 per lane)
  const f16* Qp = q + base + (size_t)(q0 + 16 * w) * HDc;
  const v16h qa0 = frag_a16(Qp, HDc);
  const v16h qa1 = frag_a16(Qp + 32, HDc);

  v8f acc[4] = {};
  float mrow[8], lrow[8];
#pragma unroll
  for (int i = 0; i < 8; ++i) { mrow[i] = -1e30f; lrow[i] = 0.f; }

  for (int c0 = 0; c0 < Sc; c0 += 64) {
    __syncthreads();
#pragma unroll
    for (int idx = tid; idx < 512; idx += 128) {            // 64 rows x 8 chunks each
      const int r = idx >> 3, c = (idx & 7) * 8;
      *(v8h*)&Ks[r * 72 + c] = *(const v8h*)&k[base + (size_t)(c0 + r) * HDc + c];
      *(v8h*)&Vt[r * 72 + c] = *(const v8h*)&v[baseV + (size_t)r * Sc + c0 + c];
    }
    __syncthreads();

    // scores: S = Q K^T * scale (4 tiles of 16 keys, 2 WMMAs each; Ks is [n][k] here)
    v8f s[4];
#pragma unroll
    for (int t = 0; t < 4; ++t) {
      const v16h b0 = frag_bt16(&Ks[(16 * t) * 72], 72);        // hd 0..31
      const v16h b1 = frag_bt16(&Ks[(16 * t) * 72 + 32], 72);   // hd 32..63
      v8f sv = {};
      sv = WMMA(qa0, b0, sv);
      sv = WMMA(qa1, b1, sv);
#pragma unroll
      for (int i = 0; i < 8; ++i) sv[i] *= SCALEc;
      s[t] = sv;
    }
    // chunk row-max across the 16 lanes that share each row
    float cm[8];
#pragma unroll
    for (int i = 0; i < 8; ++i)
      cm[i] = fmaxf(fmaxf(s[0][i], s[1][i]), fmaxf(s[2][i], s[3][i]));
#pragma unroll
    for (int off = 1; off < 16; off <<= 1)
#pragma unroll
      for (int i = 0; i < 8; ++i) cm[i] = fmaxf(cm[i], __shfl_xor(cm[i], off, 32));

    float mnew[8], alpha[8], rs[8];
#pragma unroll
    for (int i = 0; i < 8; ++i) {
      mnew[i] = fmaxf(mrow[i], cm[i]);
      alpha[i] = expf(mrow[i] - mnew[i]);
      rs[i] = 0.f;
    }
#pragma unroll
    for (int t = 0; t < 4; ++t)
#pragma unroll
      for (int i = 0; i < 8; ++i) {
        const float p = expf(s[t][i] - mnew[i]);
        s[t][i] = p;
        rs[i] += p;
      }
#pragma unroll
    for (int off = 1; off < 16; off <<= 1)
#pragma unroll
      for (int i = 0; i < 8; ++i) rs[i] += __shfl_xor(rs[i], off, 32);
#pragma unroll
    for (int i = 0; i < 8; ++i) {
      lrow[i] = lrow[i] * alpha[i] + rs[i];
      mrow[i] = mnew[i];
    }
#pragma unroll
    for (int t = 0; t < 4; ++t)
#pragma unroll
      for (int i = 0; i < 8; ++i) acc[t][i] *= alpha[i];

    // stage P (wave-private LDS, row-major [m][key]) then re-read as A fragments (b128)
#pragma unroll
    for (int t = 0; t < 4; ++t)
#pragma unroll
      for (int i = 0; i < 8; ++i)
        Ps[w][(8 * g + i) * 64 + 16 * t + ln] = (f16)s[t][i];

    const v16h pa0 = frag_a16(&Ps[w][0], 64);    // keys 0..31
    const v16h pa1 = frag_a16(&Ps[w][32], 64);   // keys 32..63
#pragma unroll
    for (int to = 0; to < 4; ++to) {
      const v16h vb0 = frag_bt16(&Vt[(16 * to) * 72], 72);        // keys 0..31
      const v16h vb1 = frag_bt16(&Vt[(16 * to) * 72 + 32], 72);   // keys 32..63
      acc[to] = WMMA(pa0, vb0, acc[to]);
      acc[to] = WMMA(pa1, vb1, acc[to]);
    }
  }
  // normalize + store ctx as [B,S,D] with D = h*64+hd
#pragma unroll
  for (int t = 0; t < 4; ++t)
#pragma unroll
    for (int i = 0; i < 8; ++i) {
      const int row = q0 + 16 * w + 8 * g + i;
      const int hd = 16 * t + ln;
      ctx[((size_t)b * Sc + row) * Dc + h * HDc + hd] = (f16)(acc[t][i] / lrow[i]);
    }
}

// ----------------------------------- exact row-0 attention per (b,h)
__global__ __launch_bounds__(256) void attn_row0_k(const f16* __restrict__ q,
    const f16* __restrict__ k, float* __restrict__ attn0) {
  const int bh = blockIdx.x;
  const int tid = threadIdx.x;
  __shared__ float qs[64];
  __shared__ float red[256];
  const size_t base = (size_t)bh * Sc * HDc;
  if (tid < 64) qs[tid] = (float)q[base + tid];
  __syncthreads();
  float e[8];
  float mx = -1e30f;
#pragma unroll
  for (int j = 0; j < 8; ++j) {
    const int key = tid + 256 * j;
    float dot = 0.f;
    for (int d = 0; d < 64; ++d) dot += qs[d] * (float)k[base + (size_t)key * HDc + d];
    e[j] = dot * SCALEc;
    mx = fmaxf(mx, e[j]);
  }
  red[tid] = mx; __syncthreads();
  for (int st = 128; st > 0; st >>= 1) { if (tid < st) red[tid] = fmaxf(red[tid], red[tid + st]); __syncthreads(); }
  const float gm = red[0];
  __syncthreads();
  float sum = 0.f;
#pragma unroll
  for (int j = 0; j < 8; ++j) { e[j] = expf(e[j] - gm); sum += e[j]; }
  red[tid] = sum; __syncthreads();
  for (int st = 128; st > 0; st >>= 1) { if (tid < st) red[tid] += red[tid + st]; __syncthreads(); }
  const float inv = 1.f / red[0];
#pragma unroll
  for (int j = 0; j < 8; ++j)
    attn0[(size_t)bh * Sc + tid + 256 * j] = e[j] * inv;
}

__global__ __launch_bounds__(256) void avg_attn_k(const float* __restrict__ attn0,
                                                  float* __restrict__ out) {
  const int idx = blockIdx.x * 256 + threadIdx.x;   // 0..B*S-1
  const int b = idx >> 11, s = idx & (Sc - 1);
  float sum = 0.f;
#pragma unroll
  for (int h = 0; h < Hc; ++h) sum += attn0[((size_t)(b * Hc + h)) * Sc + s];
  out[idx] = sum * (1.f / Hc);
}

__global__ __launch_bounds__(256) void mean_rows_k(const float* __restrict__ outln,
                                                   float* __restrict__ gc) {
  const int idx = blockIdx.x * 256 + threadIdx.x;   // 0..B*D-1
  const int b = idx >> 10, d = idx & (Dc - 1);
  float s = 0.f;
  for (int r = 0; r < Sc; ++r) s += outln[((size_t)(b * Sc + r)) * Dc + d];
  gc[idx] = s * (1.f / Sc);
}

// ---------------------------------------------------------------- launch
extern "C" void kernel_launch(void* const* d_in, const int* in_sizes, int n_in,
                              void* d_out, int out_size, void* d_ws, size_t ws_size,
                              hipStream_t stream) {
  (void)in_sizes; (void)n_in; (void)out_size; (void)ws_size;
  const float* x  = (const float*)d_in[0];
  const float* Wq = (const float*)d_in[1];
  const float* bq = (const float*)d_in[2];
  const float* Wk = (const float*)d_in[3];
  const float* bk = (const float*)d_in[4];
  const float* Wv = (const float*)d_in[5];
  const float* bv = (const float*)d_in[6];
  const float* Wo = (const float*)d_in[7];
  const float* bo = (const float*)d_in[8];
  const float* g1 = (const float*)d_in[9];
  const float* b1 = (const float*)d_in[10];
  const float* g2 = (const float*)d_in[11];
  const float* b2 = (const float*)d_in[12];
  float* out = (float*)d_out;

  char* ws = (char*)d_ws;
  size_t off = 0;
  auto alloc = [&](size_t bytes) -> char* {
    char* p = ws + off;
    off = (off + bytes + 255) & ~(size_t)255;
    return p;
  };
  const size_t DD2 = (size_t)Dc * Dc * 2;
  const size_t BD2 = (size_t)BSc * Dc * 2;
  const size_t BD4 = (size_t)BSc * Dc * 4;
  f16* h16    = (f16*)alloc(BD2);
  f16* WqT16  = (f16*)alloc(DD2);
  f16* WkT16  = (f16*)alloc(DD2);
  f16* WvT16  = (f16*)alloc(DD2);
  f16* WoT16  = (f16*)alloc(DD2);
  f16* q16    = (f16*)alloc(BD2);
  f16* k16    = (f16*)alloc(BD2);
  f16* v16    = (f16*)alloc(BD2);   // [B,H,HD,S]
  f16* ctx16  = (f16*)alloc(BD2);
  float* proj  = (float*)alloc(BD4);
  float* outln = (float*)alloc(BD4);
  float* attn0 = (float*)alloc((size_t)Bc * Hc * Sc * 4);

  const int nW = Dc * Dc;
  cvt_t_f32_f16<<<(nW + 255) / 256, 256, 0, stream>>>(Wq, WqT16, nW);
  cvt_t_f32_f16<<<(nW + 255) / 256, 256, 0, stream>>>(Wk, WkT16, nW);
  cvt_t_f32_f16<<<(nW + 255) / 256, 256, 0, stream>>>(Wv, WvT16, nW);
  cvt_t_f32_f16<<<(nW + 255) / 256, 256, 0, stream>>>(Wo, WoT16, nW);

  layernorm_k<<<BSc, 256, 0, stream>>>(x, nullptr, g1, b1, h16, nullptr, 0);

  gemm_qkv<<<dim3(Dc / 64, BSc / 256, 3), 256, 0, stream>>>(
      h16, WqT16, WkT16, WvT16, bq, bk, bv, q16, k16, v16);

  attention_k<<<dim3(Sc / 64, Bc * Hc), 128, 0, stream>>>(q16, k16, v16, ctx16);

  attn_row0_k<<<Bc * Hc, 256, 0, stream>>>(q16, k16, attn0);
  avg_attn_k<<<(Bc * Sc) / 256, 256, 0, stream>>>(attn0, out + Bc * Dc);

  gemm_o<<<dim3(Dc / 64, BSc / 256), 256, 0, stream>>>(ctx16, WoT16, bo, proj);

  layernorm_k<<<BSc, 256, 0, stream>>>(x, proj, g2, b2, nullptr, outln, 1);

  mean_rows_k<<<(Bc * Dc) / 256, 256, 0, stream>>>(outln, out);
}
